// PraxisByteLatentEncoder_8564164789056
// MI455X (gfx1250) — compile-verified, matrix operand order checked by reading, not run on hardware
//
#include <hip/hip_runtime.h>
#include <hip/hip_bf16.h>

// ---------------- problem constants ----------------
#define BB   4
#define SS   4096
#define DD   768
#define PP   683
#define NROW (BB * SS)          // 16384 rows
#define HV   65536              // hash vocab 8192*8
#define NOUT 260
#define NPAD_OUT 384            // 260 padded to multiple of 128
#define EPSF 1e-6f

typedef __bf16 bf16t;
typedef bf16t v16bf __attribute__((ext_vector_type(16)));
typedef float v8f   __attribute__((ext_vector_type(8)));

union Frag16 { v16bf v; unsigned int u[8]; };

// fp32 -> bf16 round-to-nearest-even
__device__ __forceinline__ unsigned short f2bf(float f) {
    unsigned int u = __float_as_uint(f);
    u += 0x7FFFu + ((u >> 16) & 1u);
    return (unsigned short)(u >> 16);
}

__device__ __forceinline__ float softplusf(float x) {
    return (x > 20.f) ? x : log1pf(__expf(x));
}

__device__ __forceinline__ float logaddexpf_(float a, float b) {
    float m = fmaxf(a, b);
    if (m == -INFINITY) return -INFINITY;
    return m + log1pf(__expf(fminf(a, b) - m));
}

// ---------------- patch ids + starts (single block) ----------------
__global__ void patch_kernel(const int* __restrict__ plen,
                             int* __restrict__ pstart,   // [BB][PP+1]
                             int* __restrict__ pid) {    // [BB][SS]
    int tid = threadIdx.x;
    if (tid < BB) {
        int acc = 0;
        for (int p = 0; p < PP; ++p) {
            pstart[tid * (PP + 1) + p] = acc;
            acc += plen[tid * PP + p];
        }
        pstart[tid * (PP + 1) + PP] = acc;   // == SS
    }
    __syncthreads();
    for (int bp = tid; bp < BB * PP; bp += blockDim.x) {
        int b = bp / PP, p = bp % PP;
        int s0 = pstart[b * (PP + 1) + p];
        int s1 = pstart[b * (PP + 1) + p + 1];
        for (int s = s0; s < s1; ++s) pid[b * SS + s] = p;
    }
}

// ---------------- weight convert: W[K][N] fp32 -> Wt[Npad][K] bf16 ----------------
__global__ void wconv_kernel(const float* __restrict__ W,
                             unsigned short* __restrict__ Wt,
                             int K, int N, int Npad) {
    size_t idx = (size_t)blockIdx.x * blockDim.x + threadIdx.x;
    size_t tot = (size_t)Npad * K;
    if (idx >= tot) return;
    int k = (int)(idx % K);
    int n = (int)(idx / K);
    float v = (n < N) ? W[(size_t)k * N + n] : 0.f;
    Wt[idx] = f2bf(v);
}

// ---------------- byte + hash embedding gather ----------------
__global__ void embed_kernel(const int* __restrict__ ids,
                             const float* __restrict__ tok,
                             const float* __restrict__ hemb,
                             float* __restrict__ out) {
    size_t idx = (size_t)blockIdx.x * blockDim.x + threadIdx.x;
    if (idx >= (size_t)NROW * DD) return;
    int d = (int)(idx % DD);
    int bs = (int)(idx / DD);
    int b = bs / SS, s = bs % SS;
    int t0 = ids[bs];
    int acc = t0, p = 31;
    #pragma unroll
    for (int j = 1; j < 4; ++j) {
        int sj = s - j;
        int tj = (sj >= 0) ? ids[b * SS + sj] : 0;
        acc += tj * p;
        p *= 31;
    }
    int h = acc & (HV - 1);
    out[idx] = tok[(size_t)t0 * DD + d] + hemb[(size_t)h * DD + d];
}

// ---------------- RMSNorm row kernel: fp32 in -> bf16 out ----------------
__global__ __launch_bounds__(256)
void rmsnorm_kernel(const float* __restrict__ x,
                    const float* __restrict__ w,
                    unsigned short* __restrict__ xn) {
    int row = blockIdx.x;
    const float* xr = x + (size_t)row * DD;
    __shared__ float red[256];
    float ss = 0.f;
    for (int d = threadIdx.x; d < DD; d += 256) { float v = xr[d]; ss += v * v; }
    red[threadIdx.x] = ss;
    __syncthreads();
    for (int o = 128; o > 0; o >>= 1) {
        if (threadIdx.x < o) red[threadIdx.x] += red[threadIdx.x + o];
        __syncthreads();
    }
    float scale = rsqrtf(red[0] / (float)DD + EPSF);
    for (int d = threadIdx.x; d < DD; d += 256)
        xn[(size_t)row * DD + d] = f2bf(xr[d] * scale * w[d]);
}

// ---------------- bf16 WMMA GEMM: C = A[M,K] * Bt[Npad,K]^T (+R) ----------------
// Block: 256 threads = 8 waves. Tile: 128(M) x 128(N) x 64(K), double-buffered LDS.
// Staging uses GLOBAL_LOAD_ASYNC_TO_LDS_B128 (ASYNCcnt) so the next K-panel DMA
// overlaps with WMMA compute on the current panel; one barrier per K-panel.
// Wave w owns a 32x64 sub-tile: rows [bm + (w>>1)*32, +32), cols [bn + (w&1)*64, +64).
#define GBM 128
#define GBN 128
#define GBK 64
#define LDK 72   // padded LDS K-stride (16B-aligned rows, bank decorrelation)
__global__ __launch_bounds__(256)
void gemm_bf16_kernel(const unsigned short* __restrict__ A,   // [M,K] bf16 row-major
                      const unsigned short* __restrict__ Bt,  // [Npad,K] bf16 row-major
                      const float* __restrict__ R,            // residual [M,ldC] or null
                      float* __restrict__ C,                  // [M,ldC] fp32
                      int K, int Nout, int ldC) {
    __shared__ unsigned short As[2][GBM][LDK];
    __shared__ unsigned short Bs[2][GBN][LDK];
    const int bm = blockIdx.x * GBM;
    const int bn = blockIdx.y * GBN;
    const int tid  = threadIdx.x;
    const int wave = tid >> 5;
    const int lane = tid & 31;
    const int m = lane & 15;
    const int g = lane >> 4;
    const int wm = (wave >> 1) * 32;    // 0,32,64,96
    const int wn = (wave & 1) * 64;     // 0,64

    v8f acc[2][4];
    #pragma unroll
    for (int mi = 0; mi < 2; ++mi)
        #pragma unroll
        for (int nt = 0; nt < 4; ++nt)
            acc[mi][nt] = (v8f){0.f,0.f,0.f,0.f,0.f,0.f,0.f,0.f};

    const int lr = tid >> 1;            // 0..127
    const int lc = (tid & 1) * 32;      // 0 or 32
    const unsigned short* gA = A  + (size_t)(bm + lr) * K + lc;
    const unsigned short* gB = Bt + (size_t)(bn + lr) * K + lc;

    // stage one K-panel (32 bf16 per thread per array) via async LDS DMA.
    // INST_OFFSET is added to both the LDS and global address (ISA 15.18.3),
    // so one base pair + offsets 0/16/32/48 covers the 64-byte chunk.
    auto stage = [&](int buf, int k0) {
        unsigned ldsA = (unsigned)(size_t)&As[buf][lr][lc];   // low 32 bits = LDS offset
        unsigned ldsB = (unsigned)(size_t)&Bs[buf][lr][lc];
        unsigned long long pa = (unsigned long long)(size_t)(gA + k0);
        unsigned long long pb = (unsigned long long)(size_t)(gB + k0);
        asm volatile(
            "global_load_async_to_lds_b128 %0, %2, off\n\t"
            "global_load_async_to_lds_b128 %0, %2, off offset:16\n\t"
            "global_load_async_to_lds_b128 %0, %2, off offset:32\n\t"
            "global_load_async_to_lds_b128 %0, %2, off offset:48\n\t"
            "global_load_async_to_lds_b128 %1, %3, off\n\t"
            "global_load_async_to_lds_b128 %1, %3, off offset:16\n\t"
            "global_load_async_to_lds_b128 %1, %3, off offset:32\n\t"
            "global_load_async_to_lds_b128 %1, %3, off offset:48"
            :: "v"(ldsA), "v"(ldsB), "v"(pa), "v"(pb) : "memory");
    };

    stage(0, 0);
    int cur = 0;

    for (int k0 = 0; k0 < K; k0 += GBK) {
        // previous panel's asyncs are the only outstanding ones here
        asm volatile("s_wait_asynccnt 0x0" ::: "memory");
        __syncthreads();
        if (k0 + GBK < K) stage(cur ^ 1, k0 + GBK);   // overlap DMA with compute

        #pragma unroll
        for (int ks = 0; ks < GBK; ks += 32) {
            // A fragments: 16x32 bf16, ISA per-lane K pattern
            Frag16 af[2];
            #pragma unroll
            for (int mi = 0; mi < 2; ++mi)
                #pragma unroll
                for (int v = 0; v < 8; ++v) {
                    int kk = ks + ((v < 4) ? (g * 8 + 2 * v) : (16 + g * 8 + 2 * (v - 4)));
                    af[mi].u[v] = *(const unsigned int*)&As[cur][wm + mi * 16 + m][kk];
                }
            #pragma unroll
            for (int nt = 0; nt < 4; ++nt) {
                Frag16 bf;
                #pragma unroll
                for (int v = 0; v < 8; ++v) {
                    int kk = ks + ((v < 4) ? (g * 8 + 2 * v) : (16 + g * 8 + 2 * (v - 4)));
                    bf.u[v] = *(const unsigned int*)&Bs[cur][wn + nt * 16 + m][kk];
                }
                #pragma unroll
                for (int mi = 0; mi < 2; ++mi)
                    acc[mi][nt] = __builtin_amdgcn_wmma_f32_16x16x32_bf16(
                        false, af[mi].v, false, bf.v, (short)0, acc[mi][nt], false, false);
            }
        }
        cur ^= 1;
    }

    // store: C/D layout -> VGPR r holds M = r + 8*(lane>=16), N = lane&15
    #pragma unroll
    for (int mi = 0; mi < 2; ++mi) {
        #pragma unroll
        for (int nt = 0; nt < 4; ++nt) {
            int cn = bn + wn + nt * 16 + m;
            if (cn >= Nout) continue;
            #pragma unroll
            for (int r = 0; r < 8; ++r) {
                int cm = bm + wm + mi * 16 + r + 8 * g;
                float v = acc[mi][nt][r];
                if (R) v += R[(size_t)cm * ldC + cn];
                C[(size_t)cm * ldC + cn] = v;
            }
        }
    }
}

// ---------------- minGRU wave-parallel log-space scan ----------------
// one wave32 per (b,d): lanes own 128-step chunks; compose via shfl_up scan.
// Pass 1 overwrites hg in place with (log_c, log_zh) so pass 2 skips the
// transcendental recompute (hg is scratch, regenerated by GEMM each call).
__global__ __launch_bounds__(256)
void mingru_scan_kernel(float* __restrict__ hg,              // [B*S, 2D] (mutated)
                        unsigned short* __restrict__ hOut) { // [B*S, D] bf16
    int wid  = (int)(((size_t)blockIdx.x * blockDim.x + threadIdx.x) >> 5);
    int lane = threadIdx.x & 31;
    if (wid >= BB * DD) return;
    int b = wid / DD, d = wid % DD;
    const int chunk = SS / 32;                 // 128
    const size_t rowBase = (size_t)b * SS;

    // pass 1: per-lane chunk composite (A = sum log_c, Bv = chunk result from -inf)
    float Aa = 0.f, Bv = -INFINITY;
    for (int i = 0; i < chunk; ++i) {
        int s = lane * chunk + i;
        float* row = hg + (rowBase + s) * (size_t)(2 * DD);
        float hidden = row[d];
        float gate   = row[DD + d];
        float log_c  = -softplusf(gate);
        float log_zh = -softplusf(-gate) +
                       ((hidden >= 0.f) ? __logf(hidden + 0.5f) : -softplusf(-hidden));
        row[d]      = log_c;     // cache for pass 2
        row[DD + d] = log_zh;
        Aa += log_c;
        Bv = logaddexpf_(log_c + Bv, log_zh);
    }
    // inclusive wave scan with composition (A1,B1)∘(A2,B2) = (A1+A2, lae(A2+B1, B2))
    float pA = Aa, pB = Bv;
    #pragma unroll
    for (int off = 1; off < 32; off <<= 1) {
        float oA = __shfl_up(pA, (unsigned)off, 32);
        float oB = __shfl_up(pB, (unsigned)off, 32);
        if (lane >= off) {
            pB = logaddexpf_(pA + oB, pB);
            pA = oA + pA;
        }
    }
    // exclusive prefix -> incoming state for this lane's chunk
    float inB = __shfl_up(pB, 1u, 32);
    if (lane == 0) inB = -INFINITY;

    // pass 2: replay chunk from cached (log_c, log_zh), emit h = exp(log_h)
    float log_h = inB;
    for (int i = 0; i < chunk; ++i) {
        int s = lane * chunk + i;
        const float* row = hg + (rowBase + s) * (size_t)(2 * DD);
        float log_c  = row[d];
        float log_zh = row[DD + d];
        log_h = logaddexpf_(log_c + log_h, log_zh);
        hOut[(rowBase + s) * DD + d] = f2bf(__expf(log_h));
    }
}

// ---------------- per-patch segment max (contiguous ranges, no atomics) ----------------
__global__ __launch_bounds__(256)
void segmax_kernel(const float* __restrict__ henc,
                   const int* __restrict__ pstart,
                   float* __restrict__ hp) {               // [BB][PP][DD]
    int bp = blockIdx.x;
    int b = bp / PP, p = bp % PP;
    int s0 = pstart[b * (PP + 1) + p];
    int s1 = pstart[b * (PP + 1) + p + 1];
    for (int d = threadIdx.x; d < DD; d += 256) {
        float mx = -INFINITY;
        for (int s = s0; s < s1; ++s)
            mx = fmaxf(mx, henc[((size_t)b * SS + s) * DD + d]);
        hp[((size_t)b * PP + p) * DD + d] = mx;
    }
}

// ---------------- gather patch embed back + add ----------------
__global__ void gather_add_kernel(const float* __restrict__ henc,
                                  const float* __restrict__ hp,
                                  const int* __restrict__ pid,
                                  float* __restrict__ y) {
    size_t idx = (size_t)blockIdx.x * blockDim.x + threadIdx.x;
    if (idx >= (size_t)NROW * DD) return;
    int d = (int)(idx % DD);
    int bs = (int)(idx / DD);
    int b = bs / SS;
    int p = pid[bs];
    y[idx] = henc[idx] + hp[((size_t)b * PP + p) * DD + d];
}

// ---------------- host orchestration ----------------
extern "C" void kernel_launch(void* const* d_in, const int* in_sizes, int n_in,
                              void* d_out, int out_size, void* d_ws, size_t ws_size,
                              hipStream_t stream) {
    const int*   input_ids   = (const int*)  d_in[0];
    const int*   patch_len   = (const int*)  d_in[1];
    const float* tok_emb     = (const float*)d_in[2];
    const float* hash_emb    = (const float*)d_in[3];
    const float* enc_norm_w  = (const float*)d_in[4];
    const float* enc_w_hg    = (const float*)d_in[5];
    const float* enc_w_out   = (const float*)d_in[6];
    const float* dec_norm_w  = (const float*)d_in[7];
    const float* dec_w_hg    = (const float*)d_in[8];
    const float* dec_w_out   = (const float*)d_in[9];
    const float* final_norm_w= (const float*)d_in[10];
    const float* out_proj    = (const float*)d_in[11];
    float* logits = (float*)d_out;

    // workspace carve-out
    char* ws = (char*)d_ws;
    size_t off = 0;
    auto carve = [&](size_t bytes) -> char* {
        char* p = ws + off;
        off += (bytes + 255) & ~(size_t)255;
        return p;
    };
    float*          x      = (float*)         carve((size_t)NROW * DD * 4);       // embeds / y
    float*          henc   = (float*)         carve((size_t)NROW * DD * 4);       // h_encoder / y2
    float*          hg     = (float*)         carve((size_t)NROW * 2 * DD * 4);   // gemm1 out (scratch)
    unsigned short* xn     = (unsigned short*)carve((size_t)NROW * DD * 2);       // norm out bf16
    unsigned short* hbf    = (unsigned short*)carve((size_t)NROW * DD * 2);       // scan out bf16
    float*          hp     = (float*)         carve((size_t)BB * PP * DD * 4);
    int*            pid    = (int*)           carve((size_t)BB * SS * 4);
    int*            pstart = (int*)           carve((size_t)BB * (PP + 1) * 4);
    unsigned short* whg_e  = (unsigned short*)carve((size_t)2 * DD * DD * 2);     // [1536][768]
    unsigned short* wout_e = (unsigned short*)carve((size_t)DD * DD * 2);
    unsigned short* whg_d  = (unsigned short*)carve((size_t)2 * DD * DD * 2);
    unsigned short* wout_d = (unsigned short*)carve((size_t)DD * DD * 2);
    unsigned short* wop    = (unsigned short*)carve((size_t)NPAD_OUT * DD * 2);
    (void)ws_size; (void)n_in; (void)in_sizes; (void)out_size;

    // 0) patch bookkeeping
    patch_kernel<<<1, 256, 0, stream>>>(patch_len, pstart, pid);

    // 1) weights -> transposed bf16
    auto wgrid = [](size_t n) { return (unsigned)((n + 255) / 256); };
    wconv_kernel<<<wgrid((size_t)2*DD*DD), 256, 0, stream>>>(enc_w_hg,  whg_e,  DD, 2*DD, 2*DD);
    wconv_kernel<<<wgrid((size_t)DD*DD),   256, 0, stream>>>(enc_w_out, wout_e, DD, DD,   DD);
    wconv_kernel<<<wgrid((size_t)2*DD*DD), 256, 0, stream>>>(dec_w_hg,  whg_d,  DD, 2*DD, 2*DD);
    wconv_kernel<<<wgrid((size_t)DD*DD),   256, 0, stream>>>(dec_w_out, wout_d, DD, DD,   DD);
    wconv_kernel<<<wgrid((size_t)NPAD_OUT*DD), 256, 0, stream>>>(out_proj, wop, DD, NOUT, NPAD_OUT);

    // 2) embeddings
    embed_kernel<<<wgrid((size_t)NROW * DD), 256, 0, stream>>>(input_ids, tok_emb, hash_emb, x);

    const unsigned scanBlocks = (unsigned)((BB * DD * 32) / 256);   // 384

    // 3) encoder block
    rmsnorm_kernel<<<NROW, 256, 0, stream>>>(x, enc_norm_w, xn);
    gemm_bf16_kernel<<<dim3(NROW/GBM, (2*DD)/GBN), 256, 0, stream>>>(xn, whg_e, nullptr, hg, DD, 2*DD, 2*DD);
    mingru_scan_kernel<<<scanBlocks, 256, 0, stream>>>(hg, hbf);
    gemm_bf16_kernel<<<dim3(NROW/GBM, DD/GBN), 256, 0, stream>>>(hbf, wout_e, x, henc, DD, DD, DD);

    // 4) patch pooling + gather back (y written over x; embeds dead)
    segmax_kernel<<<BB * PP, 256, 0, stream>>>(henc, pstart, hp);
    gather_add_kernel<<<wgrid((size_t)NROW * DD), 256, 0, stream>>>(henc, hp, pid, x);

    // 5) decoder block (y = x buffer; result y2 over henc)
    rmsnorm_kernel<<<NROW, 256, 0, stream>>>(x, dec_norm_w, xn);
    gemm_bf16_kernel<<<dim3(NROW/GBM, (2*DD)/GBN), 256, 0, stream>>>(xn, whg_d, nullptr, hg, DD, 2*DD, 2*DD);
    mingru_scan_kernel<<<scanBlocks, 256, 0, stream>>>(hg, hbf);
    gemm_bf16_kernel<<<dim3(NROW/GBM, DD/GBN), 256, 0, stream>>>(hbf, wout_d, x, henc, DD, DD, DD);

    // 6) final norm + output projection -> d_out [NROW, 260]
    rmsnorm_kernel<<<NROW, 256, 0, stream>>>(henc, final_norm_w, xn);
    gemm_bf16_kernel<<<dim3(NROW/GBM, NPAD_OUT/GBN), 256, 0, stream>>>(xn, wop, nullptr, logits, DD, NOUT, NOUT);
}
